// TGN_10840497455789
// MI455X (gfx1250) — matrix operand order repaired and checked by reading, not compiled, not run on hardware
//
#include <hip/hip_runtime.h>
#include <hip/hip_bf16.h>
#include <math.h>

typedef __attribute__((ext_vector_type(16))) __bf16 v16bf;
typedef __attribute__((ext_vector_type(8)))  float  v8f;

#define FEAT 64

// ---------------------------------------------------------------------------
// Degree / normalization kernels
// ---------------------------------------------------------------------------
__global__ void gcn_deg_init(float* __restrict__ deg, int n) {
    int i = blockIdx.x * blockDim.x + threadIdx.x;
    if (i < n) deg[i] = 1.0f;  // self-loop contributes 1
}

__global__ void gcn_deg_count(const int* __restrict__ dst,
                              float* __restrict__ deg, int E) {
    int e = blockIdx.x * blockDim.x + threadIdx.x;
    if (e < E) atomicAdd(&deg[dst[e]], 1.0f);
}

__global__ void gcn_deg_finalize(float* __restrict__ deg, int n) {
    int i = blockIdx.x * blockDim.x + threadIdx.x;
    if (i < n) deg[i] = rsqrtf(deg[i]);  // deg >= 1 always (self-loop)
}

// ---------------------------------------------------------------------------
// WMMA GEMM: H[n_rows, 64] = act(X[n_rows, K_DIM]) @ W[K_DIM, 64]
// Block = 128 threads = 4 waves; wave w owns the 16x16 tile at cols [16w,16w+16).
//
// A fragment (16-bit A 16x32, ISA 7.12.2): lane half h, row m=lane&15:
//   elements 0..7  <- K = 8h+0..7        (contiguous 8 floats)
//   elements 8..15 <- K = 16+8h+0..7     (contiguous 8 floats)
// -> loaded as 4 unconditional float4 (global_load_b128), row clamped in-bounds.
//
// B fragment: lane = column, element e <- same K as A (8h+e; e>=8: 16+8h+e-8). W is staged in LDS
// pre-permuted into fragment order so each lane reads 32 contiguous bytes
// per chunk (2x ds_load_b128).
// ---------------------------------------------------------------------------
template <int K_DIM, bool RELU>
__global__ void gcn_gemm_wmma(const float* __restrict__ X,
                              const float* __restrict__ W,
                              float* __restrict__ H, int n_rows) {
    constexpr int NC = K_DIM / 32;              // k-chunks of 32
    __shared__ __bf16 sB[K_DIM * FEAT];         // permuted: [(w*NC+c)*32+l][16]

    const int tid = threadIdx.x;

    // Stage W into LDS in B-fragment order. Iterate over W linearly
    // (coalesced global reads), scatter into LDS.
    for (int i = tid; i < K_DIM * FEAT; i += 128) {
        const int kk   = i >> 6;          // K index 0..K_DIM-1
        const int col  = i & 63;          // N index 0..63
        const int w    = col >> 4;        // owning wave
        const int c    = kk >> 5;         // chunk
        const int half = (kk >> 3) & 1;   // lane half
        const int e    = (kk & 7) | ((kk >> 1) & 8);  // fragment element
        const int l    = half * 16 + (col & 15);
        sB[(((w * NC + c) * 32) + l) * 16 + e] = (__bf16)W[i];
    }
    __syncthreads();

    const int wave = tid >> 5;
    const int lane = tid & 31;
    const int m    = lane & 15;
    const int half = lane >> 4;
    const int row0 = blockIdx.x * 16;
    const int col0 = wave * 16;

    // Clamp row so all loads are unconditionally in-bounds (no exec-predicated
    // scalar loads); clamped rows produce junk that is discarded at the store.
    const int rowA = (row0 + m < n_rows) ? (row0 + m) : (n_rows - 1);
    const float4* xrow = (const float4*)(X + (size_t)rowA * K_DIM);

    v8f acc = {};

    #pragma unroll
    for (int c = 0; c < NC; ++c) {
        // ---- A fragment: two contiguous 8-float runs -> 4x float4 ----
        const float4* xp = xrow + c * 8;       // 32 floats per chunk
        float4 r0 = xp[2 * half];              // K = 8h + 0..3
        float4 r1 = xp[2 * half + 1];          // K = 8h + 4..7
        float4 r2 = xp[4 + 2 * half];          // K = 16 + 8h + 0..3
        float4 r3 = xp[4 + 2 * half + 1];      // K = 16 + 8h + 4..7
        if (RELU) {
            r0.x = fmaxf(r0.x, 0.f); r0.y = fmaxf(r0.y, 0.f);
            r0.z = fmaxf(r0.z, 0.f); r0.w = fmaxf(r0.w, 0.f);
            r1.x = fmaxf(r1.x, 0.f); r1.y = fmaxf(r1.y, 0.f);
            r1.z = fmaxf(r1.z, 0.f); r1.w = fmaxf(r1.w, 0.f);
            r2.x = fmaxf(r2.x, 0.f); r2.y = fmaxf(r2.y, 0.f);
            r2.z = fmaxf(r2.z, 0.f); r2.w = fmaxf(r2.w, 0.f);
            r3.x = fmaxf(r3.x, 0.f); r3.y = fmaxf(r3.y, 0.f);
            r3.z = fmaxf(r3.z, 0.f); r3.w = fmaxf(r3.w, 0.f);
        }
        v16bf a;
        a[0]  = (__bf16)r0.x; a[1]  = (__bf16)r0.y;
        a[2]  = (__bf16)r0.z; a[3]  = (__bf16)r0.w;
        a[4]  = (__bf16)r1.x; a[5]  = (__bf16)r1.y;
        a[6]  = (__bf16)r1.z; a[7]  = (__bf16)r1.w;
        a[8]  = (__bf16)r2.x; a[9]  = (__bf16)r2.y;
        a[10] = (__bf16)r2.z; a[11] = (__bf16)r2.w;
        a[12] = (__bf16)r3.x; a[13] = (__bf16)r3.y;
        a[14] = (__bf16)r3.z; a[15] = (__bf16)r3.w;

        // ---- B fragment: 32 contiguous bytes from pre-swizzled LDS ----
        const v16bf b =
            *(const v16bf*)&sB[(((wave * NC + c) * 32) + lane) * 16];

        acc = __builtin_amdgcn_wmma_f32_16x16x32_bf16(
            /*neg_a=*/false, a, /*neg_b=*/false, b,
            /*c_mod=*/(short)0, acc, /*reuse_a=*/false, /*reuse_b=*/false);
    }

    // C layout: VGPR r -> M = r + 8*half, N = lane&15
    #pragma unroll
    for (int r = 0; r < 8; ++r) {
        const int rr = row0 + r + 8 * half;
        if (rr < n_rows)
            H[(size_t)rr * FEAT + col0 + m] = acc[r];
    }
}

// ---------------------------------------------------------------------------
// Aggregation init: out[i][f] = h[i][f] * dinv[i]^2 + bias[f]   (self-loop)
// ---------------------------------------------------------------------------
__global__ void gcn_agg_init(const float* __restrict__ h,
                             const float* __restrict__ dinv,
                             const float* __restrict__ bias,
                             float* __restrict__ out, int n) {
    long long g = (long long)blockIdx.x * blockDim.x + threadIdx.x;
    if (g >= (long long)n * FEAT) return;
    int i = (int)(g >> 6);
    int f = (int)(g & (FEAT - 1));
    float di = dinv[i];
    out[g] = h[g] * di * di + bias[f];
}

// ---------------------------------------------------------------------------
// Edge aggregation: out[dst] += h[src] * dinv[src] * dinv[dst]
// 16 threads per edge, float4 per thread (64 feats). Gathers hit the 192MB L2
// (whole h matrix is 25.6MB); atomics are L2-resident fp32 RMWs.
// ---------------------------------------------------------------------------
__global__ void gcn_edge_agg(const int* __restrict__ src,
                             const int* __restrict__ dst,
                             const float* __restrict__ h,
                             const float* __restrict__ dinv,
                             float* __restrict__ out, int E) {
    long long t = (long long)blockIdx.x * blockDim.x + threadIdx.x;
    long long e = t >> 4;
    if (e >= E) return;
    const int f4 = ((int)t & 15) * 4;
    const int s = src[e];
    const int d = dst[e];
    const float norm = dinv[s] * dinv[d];
    const float4 hv = *(const float4*)(h + (size_t)s * FEAT + f4);
    float* op = out + (size_t)d * FEAT + f4;
    atomicAdd(op + 0, hv.x * norm);
    atomicAdd(op + 1, hv.y * norm);
    atomicAdd(op + 2, hv.z * norm);
    atomicAdd(op + 3, hv.w * norm);
}

// ---------------------------------------------------------------------------
extern "C" void kernel_launch(void* const* d_in, const int* in_sizes, int n_in,
                              void* d_out, int out_size, void* d_ws, size_t ws_size,
                              hipStream_t stream) {
    const float* x   = (const float*)d_in[0];
    const int*   ei  = (const int*)d_in[1];
    const float* W1  = (const float*)d_in[2];
    const float* b1  = (const float*)d_in[3];
    const float* W2  = (const float*)d_in[4];
    const float* b2  = (const float*)d_in[5];
    float*       out = (float*)d_out;

    const int N = in_sizes[0] / 128;     // 100000
    const int E = in_sizes[1] / 2;       // 3.2M
    const int* src = ei;
    const int* dst = ei + E;

    // Workspace layout (floats): dinv[N] | h1[N*64] | a1[N*64] | h2[N*64]
    float* dinv = (float*)d_ws;
    float* h1   = dinv + N;
    float* a1   = h1 + (size_t)N * FEAT;
    float* h2   = a1 + (size_t)N * FEAT;

    const int BT = 256;
    const int nBlkN   = (N + BT - 1) / BT;
    const int nBlkE   = (E + BT - 1) / BT;
    const int nBlkNF  = (int)(((long long)N * FEAT + BT - 1) / BT);
    const int nBlkEF  = (int)(((long long)E * 16 + BT - 1) / BT);
    const int nBlkGemm = (N + 15) / 16;

    // 1. Symmetric normalization coefficients
    gcn_deg_init<<<nBlkN, BT, 0, stream>>>(dinv, N);
    gcn_deg_count<<<nBlkE, BT, 0, stream>>>(dst, dinv, E);
    gcn_deg_finalize<<<nBlkN, BT, 0, stream>>>(dinv, N);

    // 2. Layer 1: h1 = x @ W1   (K=128, WMMA bf16)
    gcn_gemm_wmma<128, false><<<nBlkGemm, 128, 0, stream>>>(x, W1, h1, N);

    // 3. Layer 1 aggregation into a1 (self-loop + bias, then edges)
    gcn_agg_init<<<nBlkNF, BT, 0, stream>>>(h1, dinv, b1, a1, N);
    gcn_edge_agg<<<nBlkEF, BT, 0, stream>>>(src, dst, h1, dinv, a1, E);

    // 4. Layer 2: h2 = relu(a1) @ W2   (K=64, relu fused into A-frag load)
    gcn_gemm_wmma<64, true><<<nBlkGemm, 128, 0, stream>>>(a1, W2, h2, N);

    // 5. Layer 2 aggregation directly into d_out
    gcn_agg_init<<<nBlkNF, BT, 0, stream>>>(h2, dinv, b2, out, N);
    gcn_edge_agg<<<nBlkEF, BT, 0, stream>>>(src, dst, h2, dinv, out, E);
}
